// R2LAttentionPlusFFN_33457795236456
// MI455X (gfx1250) — matrix operand order repaired and checked by reading, not compiled, not run on hardware
//
#include <hip/hip_runtime.h>
#include <hip/hip_bf16.h>

// ---------------------------------------------------------------------------
// R2L attention + FFN for MI455X (gfx1250, wave32, WMMA).
// All heavy GEMMs run on v_wmma_f32_16x16x32_f16 (f16 in, f32 accum).
// w_lin is algebraically folded into the expand/MLP branches (WEL/W2L/b_tot)
// so the [*,768] intermediate never exists; everything tiles through LDS.
// Workspace layout (assumes ws_size >= ~167 MB): f16 weights (~6MB),
// cls_new (3MB), x after attention (157MB).
// ---------------------------------------------------------------------------

#define CDIM 384
#define DHID 768
#define NHEADS 12
#define HD 32
#define NTOK 50
#define NPAD 64
#define HIDM 3072
#define QKVW (3 * CDIM)     // 1152
#define NWIN 2048
#define NBATCH 32
#define NWPB 64
#define SCALE_ATTN 0.17677669529663687f

typedef __attribute__((ext_vector_type(16))) _Float16 v16h;
typedef __attribute__((ext_vector_type(8)))  _Float16 v8h;
typedef __attribute__((ext_vector_type(8)))  float    v8f;

#define WMMA_F16(a, b, c) \
  __builtin_amdgcn_wmma_f32_16x16x32_f16(false, (a), false, (b), (short)0, (c), false, false)

static __device__ __forceinline__ v8f wzero() {
  v8f z;
#pragma unroll
  for (int i = 0; i < 8; ++i) z[i] = 0.0f;
  return z;
}

// A fragment (16x32 f16). Contiguous K order matching the B fragments:
// K = 16*kg + j (j = 0..15), row = lane & 15.
static __device__ __forceinline__ v16h frag_a(const _Float16* base, int stride) {
  const int lane = threadIdx.x & 31;
  const int row = lane & 15, kg = lane >> 4;
  const _Float16* p = base + (size_t)row * stride + 16 * kg;
  v8h lo = *(const v8h*)p;
  v8h hi = *(const v8h*)(p + 8);
  v16h a;
#pragma unroll
  for (int j = 0; j < 8; ++j) { a[j] = lo[j]; a[j + 8] = hi[j]; }
  return a;
}

// B fragment (32x16 f16) where memory holds B transposed: base[n*stride + k]
// (weights stored (out,in) row-major; K contiguous). order used here: col = lane&15,
// K = 16*kg + j  (j = 0..15 halves).
static __device__ __forceinline__ v16h frag_b_t(const _Float16* base, int stride) {
  const int lane = threadIdx.x & 31;
  const int col = lane & 15, kg = lane >> 4;
  return *(const v16h*)(base + (size_t)col * stride + 16 * kg);
}

// B fragment (32x16 f16) where memory holds B as-is: base[k*stride + n].
static __device__ __forceinline__ v16h frag_b_n(const _Float16* base, int stride) {
  const int lane = threadIdx.x & 31;
  const int col = lane & 15, kg = lane >> 4;
  v16h b;
#pragma unroll
  for (int j = 0; j < 16; ++j) b[j] = base[(size_t)(16 * kg + j) * stride + col];
  return b;
}

static __device__ __forceinline__ float gelu_exact(float x) {
  return 0.5f * x * (1.0f + erff(x * 0.70710678118654752440f));
}

static __device__ __forceinline__ float red_sum32(float v) {
#pragma unroll
  for (int m = 16; m >= 1; m >>= 1) v += __shfl_xor(v, m, 32);
  return v;
}
static __device__ __forceinline__ float red_max16(float v) {
#pragma unroll
  for (int m = 8; m >= 1; m >>= 1) v = fmaxf(v, __shfl_xor(v, m, 32));
  return v;
}
static __device__ __forceinline__ float red_sum16(float v) {
#pragma unroll
  for (int m = 8; m >= 1; m >>= 1) v += __shfl_xor(v, m, 32);
  return v;
}

// ---------------------------------------------------------------------------
// Precompute kernels: f16 weight copies + w_lin folding.
// ---------------------------------------------------------------------------
__global__ void convert_weights(const float* __restrict__ wqkv,
                                const float* __restrict__ wproj,
                                const float* __restrict__ wfc1,
                                _Float16* __restrict__ wqkv16,
                                _Float16* __restrict__ wproj16,
                                _Float16* __restrict__ wfc116) {
  const size_t n0 = (size_t)QKVW * CDIM, n1 = (size_t)CDIM * CDIM, n2 = (size_t)HIDM * CDIM;
  const size_t total = n0 + n1 + n2;
  for (size_t i = (size_t)blockIdx.x * blockDim.x + threadIdx.x; i < total;
       i += (size_t)gridDim.x * blockDim.x) {
    if (i < n0)            wqkv16[i] = (_Float16)wqkv[i];
    else if (i < n0 + n1)  wproj16[i - n0] = (_Float16)wproj[i - n0];
    else                   wfc116[i - n0 - n1] = (_Float16)wfc1[i - n0 - n1];
  }
}

__global__ void fuse_wel(const float* __restrict__ wlin, const float* __restrict__ wexp,
                         _Float16* __restrict__ wel16) {
  int i = blockIdx.x * blockDim.x + threadIdx.x;   // c*384 + k
  if (i >= CDIM * CDIM) return;
  int c = i / CDIM, k = i % CDIM;
  float acc = 0.0f;
  for (int d = 0; d < DHID; ++d) acc += wlin[c * DHID + d] * wexp[d * CDIM + k];
  wel16[i] = (_Float16)acc;
}

__global__ void fuse_w2l(const float* __restrict__ wlin, const float* __restrict__ wfc2,
                         _Float16* __restrict__ w2l16) {
  int i = blockIdx.x * blockDim.x + threadIdx.x;   // c*3072 + hid
  if (i >= CDIM * HIDM) return;
  int c = i / HIDM, hid = i % HIDM;
  float acc = 0.0f;
  for (int d = 0; d < DHID; ++d) acc += wlin[c * DHID + d] * wfc2[d * HIDM + hid];
  w2l16[i] = (_Float16)acc;
}

__global__ void fuse_bias(const float* __restrict__ wlin, const float* __restrict__ bexp,
                          const float* __restrict__ bfc2, const float* __restrict__ blin,
                          float* __restrict__ btot) {
  int c = blockIdx.x * blockDim.x + threadIdx.x;
  if (c >= CDIM) return;
  float acc = blin[c];
  for (int d = 0; d < DHID; ++d) acc += wlin[c * DHID + d] * (bexp[d] + bfc2[d]);
  btot[c] = acc;
}

// ---------------------------------------------------------------------------
// Fused attention kernel.
//  mode 0: cls-token attention. blockIdx = batch (32), 64 valid tokens.
//  mode 1: windowed attention. blockIdx = window (2048), 50 valid tokens,
//          rel-pos bias + mask, row 0 sourced from updated cls.
// LDS: qkv f16 [64][1152] | xln->y f16 [64][384] | per-wave P f16 [64][64].
// ---------------------------------------------------------------------------
__global__ __launch_bounds__(256) void attn_kernel(
    const float* __restrict__ out_in, const float* __restrict__ cls_in,
    const float* __restrict__ gamma, const float* __restrict__ beta,
    const _Float16* __restrict__ wqkv16, const _Float16* __restrict__ wproj16,
    const float* __restrict__ b_proj, const float* __restrict__ rel_pos,
    const float* __restrict__ mask, float* __restrict__ dst, int mode) {
  extern __shared__ char smem_raw[];
  _Float16* qkv = (_Float16*)smem_raw;            // 64*1152
  _Float16* xln = qkv + NPAD * QKVW;              // 64*384 (reused as y)
  _Float16* pbase = xln + NPAD * CDIM;            // 8 * 64*64

  const int tid = threadIdx.x;
  const int lane = tid & 31;
  const int wave = tid >> 5;
  const int blk = blockIdx.x;
  const int nvalid = (mode == 0) ? NPAD : NTOK;
  const int col_l = lane & 15, kgrp = lane >> 4;

  // ---- Phase 1: LayerNorm -> f16 LDS (pad rows zero) ----
  for (int t = wave; t < NPAD; t += 8) {
    const float* xr = nullptr;
    if (mode == 0)            xr = out_in + (size_t)(blk * NWPB + t) * NTOK * CDIM;
    else if (t == 0)          xr = cls_in + (size_t)blk * CDIM;
    else if (t < NTOK)        xr = out_in + ((size_t)blk * NTOK + t) * CDIM;
    if (xr) {
      float vals[12], s = 0.0f, s2 = 0.0f;
#pragma unroll
      for (int i = 0; i < 12; ++i) {
        float v = xr[lane + 32 * i];
        vals[i] = v; s += v; s2 += v * v;
      }
      s = red_sum32(s); s2 = red_sum32(s2);
      float mean = s * (1.0f / CDIM);
      float rstd = rsqrtf(s2 * (1.0f / CDIM) - mean * mean + 1e-5f);
#pragma unroll
      for (int i = 0; i < 12; ++i) {
        int c = lane + 32 * i;
        xln[t * CDIM + c] = (_Float16)((vals[i] - mean) * rstd * gamma[c] + beta[c]);
      }
    } else {
#pragma unroll
      for (int i = 0; i < 12; ++i) xln[t * CDIM + lane + 32 * i] = (_Float16)0.0f;
    }
  }
  __syncthreads();

  // ---- Phase 2: qkv = xln @ w_qkv.T  [64x384]x[384x1152] ----
  for (int job = wave; job < 4 * (QKVW / 16); job += 8) {
    int mt = job / (QKVW / 16);
    int nt = job % (QKVW / 16);
    v8f acc = wzero();
#pragma unroll
    for (int ks = 0; ks < 12; ++ks) {
      v16h a = frag_a(xln + mt * 16 * CDIM + ks * 32, CDIM);
      v16h b = frag_b_t(wqkv16 + (size_t)(nt * 16) * CDIM + ks * 32, CDIM);
      acc = WMMA_F16(a, b, acc);
    }
#pragma unroll
    for (int r = 0; r < 8; ++r)
      qkv[(size_t)(mt * 16 + r + 8 * kgrp) * QKVW + nt * 16 + col_l] = (_Float16)acc[r];
  }
  __syncthreads();

  // ---- Phase 3: per-head attention (y overwrites the xln region) ----
  _Float16* ylds = xln;
  _Float16* P = pbase + (size_t)wave * NPAD * NPAD;
  for (int h = wave; h < NHEADS; h += 8) {
    for (int mt = 0; mt < 4; ++mt) {
      v8f s[4];
#pragma unroll
      for (int nt = 0; nt < 4; ++nt) {
        v16h a = frag_a(qkv + (size_t)(mt * 16) * QKVW + h * HD, QKVW);
        v16h b = frag_b_t(qkv + (size_t)(nt * 16) * QKVW + CDIM + h * HD, QKVW);
        s[nt] = WMMA_F16(a, b, wzero());
      }
      // scale + rel-pos bias + mask + key-padding, then wave-local softmax
#pragma unroll
      for (int r = 0; r < 8; ++r) {
        int row = mt * 16 + r + 8 * kgrp;          // query token
        float mx = -3.0e38f;
#pragma unroll
        for (int nt = 0; nt < 4; ++nt) {
          int colk = nt * 16 + col_l;              // key token
          float v = s[nt][r] * SCALE_ATTN;
          if (colk >= nvalid) {
            v = -1.0e30f;
          } else if (mode == 1) {
            if (row >= 1 && row < NTOK && colk >= 1) {
              int qi = row - 1, ki = colk - 1;
              int idx = ((qi / 7) - (ki / 7) + 6) * 13 + ((qi % 7) - (ki % 7) + 6);
              v += rel_pos[idx * NHEADS + h];
            }
            if (row < NTOK) v += mask[((size_t)blk * NTOK + row) * NTOK + colk];
          }
          s[nt][r] = v;
          mx = fmaxf(mx, v);
        }
        mx = red_max16(mx);
        float sum = 0.0f;
#pragma unroll
        for (int nt = 0; nt < 4; ++nt) {
          float e = __expf(s[nt][r] - mx);
          s[nt][r] = e; sum += e;
        }
        sum = red_sum16(sum);
        float inv = 1.0f / sum;
#pragma unroll
        for (int nt = 0; nt < 4; ++nt)
          P[(size_t)row * NPAD + nt * 16 + col_l] = (_Float16)(s[nt][r] * inv);
      }
    }
    // y_head = P @ V   [64x64] x [64x32]
    for (int mt = 0; mt < 4; ++mt) {
#pragma unroll
      for (int nt = 0; nt < 2; ++nt) {
        v8f acc = wzero();
#pragma unroll
        for (int ks = 0; ks < 2; ++ks) {
          v16h a = frag_a(P + (size_t)(mt * 16) * NPAD + ks * 32, NPAD);
          v16h b = frag_b_n(qkv + (size_t)(ks * 32) * QKVW + 2 * CDIM + h * HD + nt * 16, QKVW);
          acc = WMMA_F16(a, b, acc);
        }
#pragma unroll
        for (int r = 0; r < 8; ++r)
          ylds[(size_t)(mt * 16 + r + 8 * kgrp) * CDIM + h * HD + nt * 16 + col_l] =
              (_Float16)acc[r];
      }
    }
  }
  __syncthreads();

  // ---- Phase 4: x = resid + y @ w_proj.T + b_proj ----
  for (int job = wave; job < 4 * 24; job += 8) {
    int mt = job / 24, nt = job % 24;
    v8f acc = wzero();
#pragma unroll
    for (int ks = 0; ks < 12; ++ks) {
      v16h a = frag_a(ylds + (size_t)(mt * 16) * CDIM + ks * 32, CDIM);
      v16h b = frag_b_t(wproj16 + (size_t)(nt * 16) * CDIM + ks * 32, CDIM);
      acc = WMMA_F16(a, b, acc);
    }
#pragma unroll
    for (int r = 0; r < 8; ++r) {
      int row = mt * 16 + r + 8 * kgrp;
      if (row < nvalid) {
        int cc = nt * 16 + col_l;
        float resid;
        float* pdst;
        if (mode == 0) {
          resid = out_in[(size_t)(blk * NWPB + row) * NTOK * CDIM + cc];
          pdst = dst + (size_t)(blk * NWPB + row) * CDIM + cc;
        } else {
          resid = (row == 0) ? cls_in[(size_t)blk * CDIM + cc]
                             : out_in[((size_t)blk * NTOK + row) * CDIM + cc];
          pdst = dst + ((size_t)blk * NTOK + row) * CDIM + cc;
        }
        *pdst = resid + acc[r] + b_proj[cc];
      }
    }
  }
}

// ---------------------------------------------------------------------------
// Fused FFN: one LN pass feeds both branches (shared moments), hidden dim
// tiled in chunks of 384 so the GELU(fc1) intermediate lives in 49 KB of LDS,
// output accumulated in 12 register C-fragments per wave against the folded
// W2L/WEL weights. 1600 blocks x 256 threads.
// ---------------------------------------------------------------------------
__global__ __launch_bounds__(256) void ffn_kernel(
    const float* __restrict__ x, const _Float16* __restrict__ wfc1,
    const float* __restrict__ bfc1, const _Float16* __restrict__ w2l,
    const _Float16* __restrict__ wel, const float* __restrict__ ge,
    const float* __restrict__ be, const float* __restrict__ g2,
    const float* __restrict__ b2, const float* __restrict__ btot,
    float* __restrict__ y) {
  extern __shared__ char smem_raw[];
  _Float16* a_exp = (_Float16*)smem_raw;        // 64*384
  _Float16* a_mlp = a_exp + NPAD * CDIM;        // 64*384
  _Float16* mchk  = a_mlp + NPAD * CDIM;        // 64*384

  const int tid = threadIdx.x, lane = tid & 31, wave = tid >> 5;
  const size_t tok0 = (size_t)blockIdx.x * NPAD;
  const int col_l = lane & 15, kgrp = lane >> 4;

  // ---- Phase 1: shared-moment LayerNorm -> both activations ----
  for (int t = wave; t < NPAD; t += 8) {
    const float* xr = x + (tok0 + t) * CDIM;
    float vals[12], s = 0.0f, s2 = 0.0f;
#pragma unroll
    for (int i = 0; i < 12; ++i) {
      float v = xr[lane + 32 * i];
      vals[i] = v; s += v; s2 += v * v;
    }
    s = red_sum32(s); s2 = red_sum32(s2);
    float mean = s * (1.0f / CDIM);
    float rstd = rsqrtf(s2 * (1.0f / CDIM) - mean * mean + 1e-5f);
#pragma unroll
    for (int i = 0; i < 12; ++i) {
      int c = lane + 32 * i;
      float nrm = (vals[i] - mean) * rstd;
      a_exp[t * CDIM + c] = (_Float16)gelu_exact(nrm * ge[c] + be[c]);
      a_mlp[t * CDIM + c] = (_Float16)(nrm * g2[c] + b2[c]);
    }
  }
  __syncthreads();

  const int mt = wave >> 1;                 // one M-tile per wave
  const int nt0 = (wave & 1) * 12;          // 12 N-tiles per wave
  v8f acc[12];
#pragma unroll
  for (int j = 0; j < 12; ++j) acc[j] = wzero();

  for (int ch = 0; ch < HIDM / CDIM; ++ch) {      // 8 chunks of 384 hidden
    // m_chunk = gelu(a_mlp @ w_fc1[chunk].T + b_fc1[chunk])
    for (int j = 0; j < 12; ++j) {
      int nt = nt0 + j;
      v8f a2 = wzero();
#pragma unroll
      for (int ks = 0; ks < 12; ++ks) {
        v16h af = frag_a(a_mlp + mt * 16 * CDIM + ks * 32, CDIM);
        v16h bf = frag_b_t(wfc1 + (size_t)(ch * CDIM + nt * 16) * CDIM + ks * 32, CDIM);
        a2 = WMMA_F16(af, bf, a2);
      }
#pragma unroll
      for (int r = 0; r < 8; ++r) {
        int hc = ch * CDIM + nt * 16 + col_l;
        mchk[(size_t)(mt * 16 + r + 8 * kgrp) * CDIM + nt * 16 + col_l] =
            (_Float16)gelu_exact(a2[r] + bfc1[hc]);
      }
    }
    __syncthreads();
    // acc += m_chunk @ W2L[:, chunk].T   (A-fragment reused across 12 N-tiles)
#pragma unroll 1
    for (int ks = 0; ks < 12; ++ks) {
      v16h af = frag_a(mchk + mt * 16 * CDIM + ks * 32, CDIM);
#pragma unroll
      for (int j = 0; j < 12; ++j) {
        v16h bf = frag_b_t(w2l + (size_t)((nt0 + j) * 16) * HIDM + ch * CDIM + ks * 32, HIDM);
        acc[j] = WMMA_F16(af, bf, acc[j]);
      }
    }
    __syncthreads();
  }

  // acc += a_exp @ WEL.T   (folded expansion path)
#pragma unroll 1
  for (int ks = 0; ks < 12; ++ks) {
    v16h af = frag_a(a_exp + mt * 16 * CDIM + ks * 32, CDIM);
#pragma unroll
    for (int j = 0; j < 12; ++j) {
      v16h bf = frag_b_t(wel + (size_t)((nt0 + j) * 16) * CDIM + ks * 32, CDIM);
      acc[j] = WMMA_F16(af, bf, acc[j]);
    }
  }

  // epilogue: + fused bias, write result
#pragma unroll
  for (int j = 0; j < 12; ++j) {
    int nt = nt0 + j;
#pragma unroll
    for (int r = 0; r < 8; ++r) {
      int row = mt * 16 + r + 8 * kgrp;
      int cc = nt * 16 + col_l;
      y[(tok0 + row) * CDIM + cc] = acc[j][r] + btot[cc];
    }
  }
}

// ---------------------------------------------------------------------------
extern "C" void kernel_launch(void* const* d_in, const int* in_sizes, int n_in,
                              void* d_out, int out_size, void* d_ws, size_t ws_size,
                              hipStream_t stream) {
  (void)in_sizes; (void)n_in; (void)out_size; (void)ws_size;
  const float* out_in = (const float*)d_in[0];
  const float* mask   = (const float*)d_in[1];
  const float* g0 = (const float*)d_in[2];
  const float* b0 = (const float*)d_in[3];
  const float* g1 = (const float*)d_in[4];
  const float* b1 = (const float*)d_in[5];
  const float* g2 = (const float*)d_in[6];
  const float* b2 = (const float*)d_in[7];
  const float* ge = (const float*)d_in[8];
  const float* be = (const float*)d_in[9];
  const float* w_qkv  = (const float*)d_in[10];
  const float* w_proj = (const float*)d_in[11];
  const float* b_proj = (const float*)d_in[12];
  const float* rel_pos = (const float*)d_in[13];
  const float* w_fc1 = (const float*)d_in[14];
  const float* b_fc1 = (const float*)d_in[15];
  const float* w_fc2 = (const float*)d_in[16];
  const float* b_fc2 = (const float*)d_in[17];
  const float* w_exp = (const float*)d_in[18];
  const float* b_exp = (const float*)d_in[19];
  const float* w_lin = (const float*)d_in[20];
  const float* b_lin = (const float*)d_in[21];

  char* ws = (char*)d_ws;
  size_t off = 0;
  auto alloc = [&](size_t bytes) -> void* {
    void* p = ws + off;
    off += bytes;
    off = (off + 255) & ~(size_t)255;
    return p;
  };
  _Float16* wqkv16  = (_Float16*)alloc((size_t)QKVW * CDIM * 2);
  _Float16* wproj16 = (_Float16*)alloc((size_t)CDIM * CDIM * 2);
  _Float16* wfc116  = (_Float16*)alloc((size_t)HIDM * CDIM * 2);
  _Float16* wel16   = (_Float16*)alloc((size_t)CDIM * CDIM * 2);
  _Float16* w2l16   = (_Float16*)alloc((size_t)CDIM * HIDM * 2);
  float* btot       = (float*)alloc((size_t)CDIM * 4);
  float* cls_new    = (float*)alloc((size_t)NWIN * CDIM * 4);
  float* x_ws       = (float*)alloc((size_t)NWIN * NTOK * CDIM * 4);

  // weight precompute (one-time per launch, trivial vs main GEMMs)
  convert_weights<<<2048, 256, 0, stream>>>(w_qkv, w_proj, w_fc1, wqkv16, wproj16, wfc116);
  fuse_wel<<<(CDIM * CDIM + 255) / 256, 256, 0, stream>>>(w_lin, w_exp, wel16);
  fuse_w2l<<<(CDIM * HIDM + 255) / 256, 256, 0, stream>>>(w_lin, w_fc2, w2l16);
  fuse_bias<<<2, 256, 0, stream>>>(w_lin, b_exp, b_fc2, b_lin, btot);

  const size_t attn_lds = (size_t)NPAD * QKVW * 2 + (size_t)NPAD * CDIM * 2 +
                          (size_t)8 * NPAD * NPAD * 2;   // 256 KB (< 320 KB WGP)
  // cls-token attention across windows of each image
  attn_kernel<<<NBATCH, 256, attn_lds, stream>>>(out_in, nullptr, g0, b0, wqkv16,
                                                 wproj16, b_proj, nullptr, nullptr,
                                                 cls_new, 0);
  // windowed patch attention with rel-pos bias + mask
  attn_kernel<<<NWIN, 256, attn_lds, stream>>>(out_in, cls_new, g1, b1, wqkv16,
                                               wproj16, b_proj, rel_pos, mask,
                                               x_ws, 1);
  // fused expand + MLP + w_lin (folded)
  const size_t ffn_lds = (size_t)3 * NPAD * CDIM * 2;    // 144 KB
  ffn_kernel<<<NWIN * NTOK / NPAD, 256, ffn_lds, stream>>>(
      x_ws, wfc116, b_fc1, w2l16, wel16, ge, be, g2, b2, btot, (float*)d_out);
}
